// LSTM_model_89721866813474
// MI455X (gfx1250) — matrix operand order repaired and checked by reading, not compiled, not run on hardware
//
#include <hip/hip_runtime.h>

typedef __attribute__((ext_vector_type(16))) _Float16 v16h;
typedef __attribute__((ext_vector_type(8)))  _Float16 v8h;
typedef __attribute__((ext_vector_type(8)))  float    v8f;

#define B_    4096
#define L_    512
#define H_    32
#define COUT_ 16

#define LOG2E_F   1.442695041f
#define TWOLOG2E_F 2.885390082f

// sigmoid(x) = 1/(1+e^-x) = rcp(1 + 2^(-x*log2e))         : mul, exp, add, rcp
__device__ __forceinline__ float sigf(float x) {
    return __builtin_amdgcn_rcpf(1.0f + __builtin_amdgcn_exp2f(x * -LOG2E_F));
}
// tanh(x) = 1 - 2/(e^(2x)+1) = 1 - 2*rcp(2^(x*2log2e) + 1) : mul, exp, add, rcp, fma
__device__ __forceinline__ float tanhfast(float x) {
    return 1.0f - 2.0f * __builtin_amdgcn_rcpf(__builtin_amdgcn_exp2f(x * TWOLOG2E_F) + 1.0f);
}

#define WMMA(A, Bt, C) \
    __builtin_amdgcn_wmma_f32_16x16x32_f16(false, (A), false, (Bt), (short)0, (C), false, false)

__global__ __launch_bounds__(64)
void lstm_fused(const float* __restrict__ x,
                const float* __restrict__ conv_w, const float* __restrict__ conv_b,
                const float* __restrict__ w_ih0,  const float* __restrict__ w_hh0,
                const float* __restrict__ b_ih0,  const float* __restrict__ b_hh0,
                const float* __restrict__ w_ih1,  const float* __restrict__ w_hh1,
                const float* __restrict__ b_ih1,  const float* __restrict__ b_hh1,
                const float* __restrict__ fc_w,   const float* __restrict__ fc_b,
                float* __restrict__ out)
{
    // One 16-row batch tile per block, 2 waves splitting the gate columns.
    // h ping-pong buffers: [t&1][layer][row m][hidden j] = 4 KB
    __shared__ __align__(16) _Float16 ldsH[2][2][16][32];

    const int lane = threadIdx.x & 31;
    const int tt   = threadIdx.x >> 5;   // 0/1 : gate-column half this wave owns
    const int n    = lane & 15;
    const int half = lane >> 4;
    const int rowBase = blockIdx.x * 16;

    // ---- B matrices (WMMA B-layout) in registers: 4 mats x 4 gate-tiles ----
    // Tile for gate g (i,f,g,o) of this wave is N-tile (2*g + tt).
    v16h Bm[4][4];
    {
        const float* Ws[4] = { w_ih0, w_hh0, w_ih1, w_hh1 };
        const int    kd[4] = { COUT_, H_, H_, H_ };
        #pragma unroll
        for (int mat = 0; mat < 4; ++mat) {
            #pragma unroll
            for (int g = 0; g < 4; ++g) {
                const int row = (2 * g + tt) * 16 + n;   // gate column index
                v16h bt;
                #pragma unroll
                for (int v = 0; v < 16; ++v) {
                    const int k = (v < 8) ? half * 8 + v : 16 + half * 8 + (v - 8);
                    const float val = (k < kd[mat]) ? Ws[mat][row * kd[mat] + k] : 0.0f;
                    bt[v] = (_Float16)val;
                }
                Bm[mat][g] = bt;
            }
        }
        // Fold layer-0 bias into the K==16 row of the zero-padded conv-B
        // (lane group half==0, element v==8 holds K=16). yA[8]=1.0 selects it.
        if (half == 0) {
            #pragma unroll
            for (int g = 0; g < 4; ++g) {
                const int j = (2 * g + tt) * 16 + n;
                Bm[0][g][8] = (_Float16)(b_ih0[j] + b_hh0[j]);
            }
        }
    }

    float bias1[4];
    #pragma unroll
    for (int g = 0; g < 4; ++g) {
        const int j = (2 * g + tt) * 16 + n;
        bias1[g] = b_ih1[j] + b_hh1[j];
    }

    // ---- conv weights per lane, packed f16 (8 channels = K indices half*8+v) ----
    v8h cw0h, cw1h, cw2h, cbh;
    #pragma unroll
    for (int v = 0; v < 8; ++v) {
        const int co = half * 8 + v;
        cw0h[v] = (_Float16)conv_w[co * 3 + 0];
        cw1h[v] = (_Float16)conv_w[co * 3 + 1];
        cw2h[v] = (_Float16)conv_w[co * 3 + 2];
        cbh[v]  = (_Float16)conv_b[co];
    }
    // Constant upper half of yA: K=16 slot carries 1.0 (bias hook), rest zero.
    const v8h yHi = { (_Float16)1.0f, (_Float16)0.0f, (_Float16)0.0f, (_Float16)0.0f,
                      (_Float16)0.0f, (_Float16)0.0f, (_Float16)0.0f, (_Float16)0.0f };

    // ---- Recurrent state ----
    v8f  c0 = {}, c1 = {};          // cell state for hidden cols tt*16..tt*16+15
    v16h h1A = {}, h2A = {};        // full hidden state, A-layout f16

    const float* xrow = x + (size_t)(rowBase + n) * L_;
    float xm  = 0.0f;
    float x0v = xrow[0];
    float xp  = xrow[1];

    for (int t = 0; t < L_; ++t) {
        const float xn = (t + 2 < L_) ? xrow[t + 2] : 0.0f;   // prefetch
        const int   buf = t & 1;

        // Conv1d in packed f16 -> A-matrix lower half (K 0..7 / 8..15 per lane group)
        const _Float16 xmh = (_Float16)xm;
        const _Float16 x0h = (_Float16)x0v;
        const _Float16 xph = (_Float16)xp;
        v8h y = cw0h * xmh + cw1h * x0h + cw2h * xph + cbh;
        const v16h yA = __builtin_shufflevector(y, yHi, 0,1,2,3,4,5,6,7,8,9,10,11,12,13,14,15);

        // ========== Layer 0 : acc[g] = bias0 + y@Wih0^T + h1@Whh0^T ==========
        v8f acc[4];
        #pragma unroll
        for (int g = 0; g < 4; ++g) {
            v8f z = {};
            z = WMMA(yA,  Bm[0][g], z);
            z = WMMA(h1A, Bm[1][g], z);
            acc[g] = z;
        }
        #pragma unroll
        for (int r = 0; r < 8; ++r) {
            const float ig = sigf(acc[0][r]);
            const float fg = sigf(acc[1][r]);
            const float gg = tanhfast(acc[2][r]);
            const float og = sigf(acc[3][r]);
            const float c  = fg * c0[r] + ig * gg;
            c0[r] = c;
            ldsH[buf][0][half * 8 + r][tt * 16 + n] = (_Float16)(og * tanhfast(c));
        }
        __syncthreads();
        {
            v8h lo = *(const v8h*)&ldsH[buf][0][n][half * 8];
            v8h hi = *(const v8h*)&ldsH[buf][0][n][16 + half * 8];
            h1A = __builtin_shufflevector(lo, hi, 0,1,2,3,4,5,6,7,8,9,10,11,12,13,14,15);
        }

        // ========== Layer 1 : acc[g] = bias1 + h1@Wih1^T + h2@Whh1^T ==========
        #pragma unroll
        for (int g = 0; g < 4; ++g) {
            const float bb = bias1[g];
            v8f z = { bb, bb, bb, bb, bb, bb, bb, bb };
            z = WMMA(h1A, Bm[2][g], z);
            z = WMMA(h2A, Bm[3][g], z);
            acc[g] = z;
        }
        #pragma unroll
        for (int r = 0; r < 8; ++r) {
            const float ig = sigf(acc[0][r]);
            const float fg = sigf(acc[1][r]);
            const float gg = tanhfast(acc[2][r]);
            const float og = sigf(acc[3][r]);
            const float c  = fg * c1[r] + ig * gg;
            c1[r] = c;
            ldsH[buf][1][half * 8 + r][tt * 16 + n] = (_Float16)(og * tanhfast(c));
        }
        __syncthreads();
        {
            v8h lo = *(const v8h*)&ldsH[buf][1][n][half * 8];
            v8h hi = *(const v8h*)&ldsH[buf][1][n][16 + half * 8];
            h2A = __builtin_shufflevector(lo, hi, 0,1,2,3,4,5,6,7,8,9,10,11,12,13,14,15);
        }

        xm = x0v; x0v = xp; xp = xn;
    }

    // ---- Final FC (32 -> 3) + softmax from h2(t=511) (buf==1 in ldsH) ----
    if (tt == 0 && lane < 16) {
        const int b = rowBase + lane;
        float l0 = fc_b[0], l1 = fc_b[1], l2 = fc_b[2];
        #pragma unroll
        for (int j = 0; j < H_; ++j) {
            const float hv = (float)ldsH[1][1][lane][j];
            l0 += hv * fc_w[0 * H_ + j];
            l1 += hv * fc_w[1 * H_ + j];
            l2 += hv * fc_w[2 * H_ + j];
        }
        const float mx = fmaxf(l0, fmaxf(l1, l2));
        const float e0 = __builtin_amdgcn_exp2f((l0 - mx) * LOG2E_F);
        const float e1 = __builtin_amdgcn_exp2f((l1 - mx) * LOG2E_F);
        const float e2 = __builtin_amdgcn_exp2f((l2 - mx) * LOG2E_F);
        const float inv = __builtin_amdgcn_rcpf(e0 + e1 + e2);
        out[b * 3 + 0] = e0 * inv;
        out[b * 3 + 1] = e1 * inv;
        out[b * 3 + 2] = e2 * inv;
    }
}

extern "C" void kernel_launch(void* const* d_in, const int* in_sizes, int n_in,
                              void* d_out, int out_size, void* d_ws, size_t ws_size,
                              hipStream_t stream) {
    const float* x      = (const float*)d_in[0];
    const float* conv_w = (const float*)d_in[1];
    const float* conv_b = (const float*)d_in[2];
    const float* w_ih0  = (const float*)d_in[3];
    const float* w_hh0  = (const float*)d_in[4];
    const float* b_ih0  = (const float*)d_in[5];
    const float* b_hh0  = (const float*)d_in[6];
    const float* w_ih1  = (const float*)d_in[7];
    const float* w_hh1  = (const float*)d_in[8];
    const float* b_ih1  = (const float*)d_in[9];
    const float* b_hh1  = (const float*)d_in[10];
    const float* fc_w   = (const float*)d_in[11];
    const float* fc_b   = (const float*)d_in[12];
    float* out = (float*)d_out;

    dim3 grid(B_ / 16);     // one 16-row tile per block, 2 waves per tile
    dim3 block(64);
    hipLaunchKernelGGL(lstm_fused, grid, block, 0, stream,
                       x, conv_w, conv_b,
                       w_ih0, w_hh0, b_ih0, b_hh0,
                       w_ih1, w_hh1, b_ih1, b_hh1,
                       fc_w, fc_b, out);
}
